// Attention_56538949484622
// MI455X (gfx1250) — matrix operand order repaired and checked by reading, not compiled, not run on hardware
//
#include <hip/hip_runtime.h>
#include <hip/hip_bf16.h>

// ---------------------------------------------------------------------------
// Types for CDNA5 WMMA (wave32)
// ---------------------------------------------------------------------------
typedef __bf16 bf16_t;
typedef __attribute__((ext_vector_type(16))) __bf16 v16bf;
typedef __attribute__((ext_vector_type(8)))  float  v8f;

#define NHEADS 32
#define NKV    8
#define SEQ    2048
#define DIM    2048
#define HD     64
#define QKVO   3072   // (32 + 2*8) * 64

__device__ __forceinline__ v8f wmma_bf16(v16bf a, v16bf b, v8f c) {
  // D = A(16x32 bf16) * B(32x16 bf16) + C(16x16 f32)
  return __builtin_amdgcn_wmma_f32_16x16x32_bf16(
      /*neg_a=*/false, a, /*neg_b=*/false, b,
      /*c_mod=*/(short)0, c, /*reuse_a=*/false, /*reuse_b=*/false);
}

// Async global->LDS 16-byte copy (ASYNCcnt tracked). Generic LDS pointer's low
// 32 bits are the LDS byte offset (aperture form {SHARED_BASE,16'b0} in MSBs).
__device__ __forceinline__ void async_b128(const bf16_t* g, bf16_t* l) {
  asm volatile("global_load_async_to_lds_b128 %0, %1, off"
               :: "v"((unsigned)(unsigned long long)l), "v"(g)
               : "memory");
}
__device__ __forceinline__ void wait_async0() {
  asm volatile("s_wait_asynccnt 0x0" ::: "memory");
}

// A-fragment: 16x32 bf16 tile, row-major source with leading dim `ld`.
// ISA layout: lane=(half,m); VGPR v holds K = (v<4 ? 2v : 16+2(v-4)) + 8*half.
__device__ __forceinline__ v16bf load_a_frag(const bf16_t* __restrict__ base, int ld) {
  int lane = threadIdx.x & 31;
  int half = lane >> 4, m = lane & 15;
  const bf16_t* row = base + (long)m * ld;
  v16bf a;
#pragma unroll
  for (int v = 0; v < 8; ++v) {
    int kk = (v < 4 ? v * 2 : 16 + (v - 4) * 2) + half * 8;
    a[2 * v]     = row[kk];
    a[2 * v + 1] = row[kk + 1];
  }
  return a;
}

// B-fragment where B[k][n] = W[n][k] (multiply by W^T), W row-major with
// leading dim ldw, K-slice at kbase. lane=(half,n); VGPR v holds K = (v<4 ? 2v : 16+2(v-4)) + 8*half.
// Per-lane reads are contiguous pairs -> 2x b128 (global) or 2x ds_load_b128.
__device__ __forceinline__ v16bf load_b_frag_wt(const bf16_t* __restrict__ w,
                                                int ldw, int kbase) {
  int lane = threadIdx.x & 31;
  int half = lane >> 4, n = lane & 15;
  const bf16_t* row = w + (long)n * ldw + kbase;
  v16bf b;
#pragma unroll
  for (int v = 0; v < 8; ++v) {
    int k = (v < 4 ? v * 2 : 16 + (v - 4) * 2) + half * 8;
    b[2 * v]     = row[k];
    b[2 * v + 1] = row[k + 1];
  }
  return b;
}

// ---------------------------------------------------------------------------
// 1a) f32 -> bf16 elementwise conversion
// ---------------------------------------------------------------------------
__global__ void k_f32_to_bf16(const float* __restrict__ s, bf16_t* __restrict__ d, int n) {
  int i = blockIdx.x * blockDim.x + threadIdx.x;
  if (i < n) d[i] = (bf16_t)s[i];
}

// ---------------------------------------------------------------------------
// 1b) Tiled transpose+convert: wo[f][d] f32 -> woT[d][f] bf16 (32x32 tiles).
// ---------------------------------------------------------------------------
__global__ __launch_bounds__(256) void k_transpose_bf16(const float* __restrict__ src,
                                                        bf16_t* __restrict__ dst) {
  __shared__ float tile[32][33];
  int r0 = blockIdx.y * 32, c0 = blockIdx.x * 32;
  for (int i = threadIdx.y; i < 32; i += 8)
    tile[i][threadIdx.x] = src[(long)(r0 + i) * DIM + c0 + threadIdx.x];
  __syncthreads();
  for (int i = threadIdx.y; i < 32; i += 8)
    dst[(long)(c0 + i) * DIM + r0 + threadIdx.x] = (bf16_t)tile[threadIdx.x][i];
}

// ---------------------------------------------------------------------------
// 2) Unified GEMM: C[M=4096][N] = A_bf16[M][K=2048] * W_bf16[N][K]^T, f32 out.
//    Block = 8 waves; each wave a 16x64 strip (same nbase across waves).
//    B tile (64 rows x 32 k = 4KB) staged in LDS by async b128 copies,
//    double-buffered; A fragments software-pipelined one step ahead; last
//    K-step peeled so the steady-state loop is branch-free.
//    grid = (N/64, M/128), block = 256.
// ---------------------------------------------------------------------------
__global__ __launch_bounds__(256) void k_gemm_wt(const bf16_t* __restrict__ A,
                                                 const bf16_t* __restrict__ W,
                                                 float* __restrict__ C, int N) {
  const int K = DIM;
  __shared__ __align__(16) bf16_t bb[2][64][32];
  int tid  = threadIdx.x;
  int wave = tid >> 5;
  int lane = tid & 31;
  int half = lane >> 4, n = lane & 15;
  int mbase = (blockIdx.y * 8 + wave) * 16;
  int nbase = blockIdx.x * 64;
  const bf16_t* Abase = A + (long)mbase * K;

  int srow = tid >> 2, sseg = (tid & 3) * 8;  // 64 rows x 4 segs of 8 bf16
  const bf16_t* wrow = W + (long)(nbase + srow) * K + sseg;

  v8f acc[4] = {};
  auto compute = [&](int bsel, v16bf a) {
    // all 4 B fragments first (one dscnt wait), then 4 WMMAs
    v16bf b0 = load_b_frag_wt(&bb[bsel][0][0],  32, 0);
    v16bf b1 = load_b_frag_wt(&bb[bsel][16][0], 32, 0);
    v16bf b2 = load_b_frag_wt(&bb[bsel][32][0], 32, 0);
    v16bf b3 = load_b_frag_wt(&bb[bsel][48][0], 32, 0);
    acc[0] = wmma_bf16(a, b0, acc[0]);
    acc[1] = wmma_bf16(a, b1, acc[1]);
    acc[2] = wmma_bf16(a, b2, acc[2]);
    acc[3] = wmma_bf16(a, b3, acc[3]);
  };

  // prologue: stage k=0 tile, preload first A fragment
  async_b128(wrow, &bb[0][srow][sseg]);
  v16bf a_cur = load_a_frag(Abase, K);
  wait_async0();
  __syncthreads();

  int buf = 0;
#pragma unroll 2
  for (int k = 0; k < K - 32; k += 32) {   // steady state: branch-free body
    async_b128(wrow + k + 32, &bb[buf ^ 1][srow][sseg]);
    __builtin_prefetch(Abase + k + 64, 0, 0);
    v16bf a_nxt = load_a_frag(Abase + k + 32, K);
    compute(buf, a_cur);
    wait_async0();
    __syncthreads();
    a_cur = a_nxt;
    buf ^= 1;
  }
  compute(buf, a_cur);                     // peeled last K-step

#pragma unroll
  for (int t = 0; t < 4; ++t)
#pragma unroll
    for (int r = 0; r < 8; ++r)
      C[(long)(mbase + r + 8 * half) * N + nbase + t * 16 + n] = acc[t][r];
}

// ---------------------------------------------------------------------------
// 3) LayerNorm + RoPE + relayout to bf16 (one wave per 64-wide row).
// ---------------------------------------------------------------------------
__global__ __launch_bounds__(256) void k_ln_rope(const float* __restrict__ xqkv,
                                                 const float* __restrict__ qg,
                                                 const float* __restrict__ qb,
                                                 const float* __restrict__ kg,
                                                 const float* __restrict__ kb,
                                                 const float* __restrict__ fc,
                                                 const float* __restrict__ fs,
                                                 bf16_t* __restrict__ Qo,
                                                 bf16_t* __restrict__ Ko,
                                                 bf16_t* __restrict__ Vo) {
  int rid  = (blockIdx.x * blockDim.x + threadIdx.x) >> 5;
  int lane = threadIdx.x & 31;
  if (rid >= 2 * SEQ * 48) return;
  int b = rid / (SEQ * 48);
  int rem = rid % (SEQ * 48);
  int s = rem / 48;
  int c = rem % 48;
  int col = (c < 32) ? c * 64 : (c < 40) ? 2048 + (c - 32) * 64 : 2560 + (c - 40) * 64;
  const float* src = xqkv + ((long)(b * SEQ + s)) * QKVO + col;
  float v0 = src[2 * lane], v1 = src[2 * lane + 1];
  if (c < 40) {
    float sum = v0 + v1, sq = v0 * v0 + v1 * v1;
#pragma unroll
    for (int m = 16; m >= 1; m >>= 1) {
      sum += __shfl_xor(sum, m, 32);
      sq  += __shfl_xor(sq,  m, 32);
    }
    float mu   = sum * (1.0f / 64.0f);
    float var  = sq * (1.0f / 64.0f) - mu * mu;
    float rstd = rsqrtf(var + 1e-5f);
    const float* g  = (c < 32) ? qg : kg;
    const float* be = (c < 32) ? qb : kb;
    v0 = (v0 - mu) * rstd * g[2 * lane]     + be[2 * lane];
    v1 = (v1 - mu) * rstd * g[2 * lane + 1] + be[2 * lane + 1];
    float co = fc[s * 32 + lane], si = fs[s * 32 + lane];
    float re = v0, im = v1;
    v0 = re * co - im * si;
    v1 = re * si + im * co;
  }
  if (c < 32) {
    bf16_t* dst = Qo + (((long)(b * NHEADS + c)) * SEQ + s) * HD + 2 * lane;
    dst[0] = (bf16_t)v0; dst[1] = (bf16_t)v1;
  } else if (c < 40) {
    bf16_t* dst = Ko + (((long)(b * NKV + (c - 32))) * SEQ + s) * HD + 2 * lane;
    dst[0] = (bf16_t)v0; dst[1] = (bf16_t)v1;
  } else {
    bf16_t* dst = Vo + (((long)(b * NKV + (c - 40))) * SEQ + s) * HD + 2 * lane;
    dst[0] = (bf16_t)v0; dst[1] = (bf16_t)v1;
  }
}

// ---------------------------------------------------------------------------
// 4) Causal flash attention, GQA 4:1.
//    Block = 8 waves (128 q rows), one kv head; UNIFORM key loop so all waves
//    share double-buffered LDS tiles: K (32x64, async b128 copies) and V
//    transposed (64x32, manual b128 load + ds scatter). Batched fragment loads
//    before each WMMA group; last key block peeled (branch-free steady state).
//    fp32 online softmax; P re-laid-out through a per-wave LDS tile.
//    grid = (S/128, H, B), block = 256.
// ---------------------------------------------------------------------------
__global__ __launch_bounds__(256) void k_attn(const bf16_t* __restrict__ Q,
                                              const bf16_t* __restrict__ Kt,
                                              const bf16_t* __restrict__ Vt,
                                              bf16_t* __restrict__ O) {
  __shared__ __align__(16) bf16_t kbuf[2][32][64];   // [key][d]
  __shared__ __align__(16) bf16_t vtbuf[2][64][32];  // [d][key] (transposed)
  __shared__ __align__(16) bf16_t pbuf[8][16 * 32];  // per-wave P tile
  int tid  = threadIdx.x;
  int wave = tid >> 5, lane = tid & 31;
  int half = lane >> 4, n = lane & 15;
  int qblock = blockIdx.x * 128;
  int qbase  = qblock + wave * 16;
  int h = blockIdx.y, b = blockIdx.z;
  int kv = h >> 2;  // N_REP = 4
  const bf16_t* qp = Q  + (((long)(b * NHEADS + h)) * SEQ + qbase) * HD;
  const bf16_t* kp = Kt + ((long)(b * NKV + kv)) * SEQ * HD;
  const bf16_t* vp = Vt + ((long)(b * NKV + kv)) * SEQ * HD;

  int skey = tid >> 3, sseg = (tid & 7) * 8;  // 32 keys x 8 segs of 8 bf16

  auto stage = [&](int d, int j) {
    async_b128(kp + (long)(j + skey) * HD + sseg, &kbuf[d][skey][sseg]);
    int4 raw = *reinterpret_cast<const int4*>(vp + (long)(j + skey) * HD + sseg);
    const bf16_t* tp = reinterpret_cast<const bf16_t*>(&raw);
#pragma unroll
    for (int i = 0; i < 8; ++i) vtbuf[d][sseg + i][skey] = tp[i];
  };

  v16bf qa0 = load_a_frag(qp, HD);       // q dims 0..31
  v16bf qa1 = load_a_frag(qp + 32, HD);  // q dims 32..63
  v8f acc[4] = {};
  float mrow[8], lrow[8];
#pragma unroll
  for (int r = 0; r < 8; ++r) { mrow[r] = -3.0e38f; lrow[r] = 0.0f; }
  const float scale = 0.125f;  // 1/sqrt(64)

  auto compute_block = [&](int j, int bsel) {
    // --- scores: batch 4 K-fragments, then 4 WMMAs ---
    v16bf b0a = load_b_frag_wt(&kbuf[bsel][0][0],  HD, 0);
    v16bf b0b = load_b_frag_wt(&kbuf[bsel][0][0],  HD, 32);
    v16bf b1a = load_b_frag_wt(&kbuf[bsel][16][0], HD, 0);
    v16bf b1b = load_b_frag_wt(&kbuf[bsel][16][0], HD, 32);
    v8f s0 = {}, s1 = {};
    s0 = wmma_bf16(qa0, b0a, s0);
    s0 = wmma_bf16(qa1, b0b, s0);
    s1 = wmma_bf16(qa0, b1a, s1);
    s1 = wmma_bf16(qa1, b1b, s1);
    // --- scale + causal mask + fp32 online softmax ---
    float p0[8], p1[8];
#pragma unroll
    for (int r = 0; r < 8; ++r) {
      int qi = qbase + r + 8 * half;
      float x0 = s0[r] * scale; if (j + n      > qi) x0 = -1.0e9f;
      float x1 = s1[r] * scale; if (j + 16 + n > qi) x1 = -1.0e9f;
      p0[r] = x0; p1[r] = x1;
    }
#pragma unroll
    for (int r = 0; r < 8; ++r) {
      float mx = fmaxf(p0[r], p1[r]);
#pragma unroll
      for (int m = 8; m >= 1; m >>= 1) mx = fmaxf(mx, __shfl_xor(mx, m, 32));
      float mnew = fmaxf(mrow[r], mx);
      float corr = __expf(mrow[r] - mnew);
      float e0 = __expf(p0[r] - mnew);
      float e1 = __expf(p1[r] - mnew);
      float rs = e0 + e1;
#pragma unroll
      for (int m = 8; m >= 1; m >>= 1) rs += __shfl_xor(rs, m, 32);
      lrow[r] = lrow[r] * corr + rs;
      mrow[r] = mnew;
#pragma unroll
      for (int t = 0; t < 4; ++t) acc[t][r] *= corr;
      p0[r] = e0; p1[r] = e1;
    }
    // --- C-tile -> A-fragment via per-wave LDS tile (wave-ordered DS) ---
    bf16_t* pb = pbuf[wave];
#pragma unroll
    for (int r = 0; r < 8; ++r) {
      int m = r + 8 * half;
      pb[m * 32 + n]      = (bf16_t)p0[r];
      pb[m * 32 + 16 + n] = (bf16_t)p1[r];
    }
    v16bf pa = load_a_frag(pb, 32);
    // --- PV: batch 4 V^T fragments, then 4 WMMAs ---
    v16bf v0f = load_b_frag_wt(&vtbuf[bsel][0][0],  32, 0);
    v16bf v1f = load_b_frag_wt(&vtbuf[bsel][16][0], 32, 0);
    v16bf v2f = load_b_frag_wt(&vtbuf[bsel][32][0], 32, 0);
    v16bf v3f = load_b_frag_wt(&vtbuf[bsel][48][0], 32, 0);
    acc[0] = wmma_bf16(pa, v0f, acc[0]);
    acc[1] = wmma_bf16(pa, v1f, acc[1]);
    acc[2] = wmma_bf16(pa, v2f, acc[2]);
    acc[3] = wmma_bf16(pa, v3f, acc[3]);
  };

  int kend = qblock + 128;  // uniform causal bound for the whole block
  stage(0, 0);
  wait_async0();
  __syncthreads();

  int buf = 0;
  for (int j = 0; j < kend - 32; j += 32) {  // steady state: branch-free
    stage(buf ^ 1, j + 32);
    compute_block(j, buf);
    wait_async0();
    __syncthreads();
    buf ^= 1;
  }
  compute_block(kend - 32, buf);             // peeled last key block

  // --- epilogue: normalize and store bf16 in (B, S, H*HD) layout ---
#pragma unroll
  for (int r = 0; r < 8; ++r) {
    float inv = 1.0f / lrow[r];
    int qi = qbase + r + 8 * half;
    bf16_t* dst = O + ((long)b * SEQ + qi) * DIM + h * HD;
#pragma unroll
    for (int t = 0; t < 4; ++t) dst[t * 16 + n] = (bf16_t)(acc[t][r] * inv);
  }
}

// ---------------------------------------------------------------------------
// Launcher
// ---------------------------------------------------------------------------
extern "C" void kernel_launch(void* const* d_in, const int* in_sizes, int n_in,
                              void* d_out, int out_size, void* d_ws, size_t ws_size,
                              hipStream_t stream) {
  const float* x    = (const float*)d_in[0];
  const float* wqkv = (const float*)d_in[1];
  const float* wo   = (const float*)d_in[2];
  const float* qg   = (const float*)d_in[3];
  const float* qb   = (const float*)d_in[4];
  const float* kg   = (const float*)d_in[5];
  const float* kb   = (const float*)d_in[6];
  const float* fc   = (const float*)d_in[7];
  const float* fs   = (const float*)d_in[8];
  // d_in[9] (mask) unused: causal mask generated in-kernel.

  char* ws = (char*)d_ws;
  const size_t n_x    = (size_t)2 * SEQ * DIM;   // 8,388,608
  const size_t n_wqkv = (size_t)QKVO * DIM;      // 6,291,456
  const size_t n_wo   = (size_t)DIM * DIM;       // 4,194,304
  bf16_t* xb    = (bf16_t*)ws;  ws += n_x * 2;
  bf16_t* wqkvb = (bf16_t*)ws;  ws += n_wqkv * 2;
  bf16_t* woT   = (bf16_t*)ws;  ws += n_wo * 2;                    // [d][f]
  float*  xqkv  = (float*)ws;   ws += (size_t)2 * SEQ * QKVO * 4;
  bf16_t* Qb    = (bf16_t*)ws;  ws += (size_t)2 * NHEADS * SEQ * HD * 2;
  bf16_t* Kb    = (bf16_t*)ws;  ws += (size_t)2 * NKV * SEQ * HD * 2;
  bf16_t* Vb    = (bf16_t*)ws;  ws += (size_t)2 * NKV * SEQ * HD * 2;
  bf16_t* Ob    = (bf16_t*)ws;  ws += (size_t)2 * SEQ * DIM * 2;

  // 1) conversions (wo transposed so both GEMMs take the W[n][k] path)
  k_f32_to_bf16<<<(int)((n_x    + 255) / 256), 256, 0, stream>>>(x,    xb,    (int)n_x);
  k_f32_to_bf16<<<(int)((n_wqkv + 255) / 256), 256, 0, stream>>>(wqkv, wqkvb, (int)n_wqkv);
  k_transpose_bf16<<<dim3(DIM / 32, DIM / 32), dim3(32, 8), 0, stream>>>(wo, woT);

  // 2) QKV GEMM: M=4096, N=3072, K=2048
  k_gemm_wt<<<dim3(QKVO / 64, (2 * SEQ) / 128), 256, 0, stream>>>(xb, wqkvb, xqkv, QKVO);

  // 3) LN + RoPE + relayout
  k_ln_rope<<<(2 * SEQ * 48) / 8, 256, 0, stream>>>(xqkv, qg, qb, kg, kb, fc, fs,
                                                    Qb, Kb, Vb);

  // 4) flash attention
  k_attn<<<dim3(SEQ / 128, NHEADS, 2), 256, 0, stream>>>(Qb, Kb, Vb, Ob);

  // 5) output GEMM: M=4096, N=2048, K=2048 -> d_out (fp32)
  k_gemm_wt<<<dim3(DIM / 64, (2 * SEQ) / 128), 256, 0, stream>>>(Ob, woT, (float*)d_out, DIM);
}
